// StandaloneAttention_81063212745020
// MI455X (gfx1250) — matrix operand order repaired and checked by reading, not compiled, not run on hardware
//
#include <hip/hip_runtime.h>
#include <hip/hip_bf16.h>

#define S_LEN 4096
#define HID_N 2048
#define HQ_N  16
#define HKV_N 8
#define DHEAD 128
#define WIN   1024

typedef __attribute__((ext_vector_type(16))) __bf16 bf16x16;
typedef __attribute__((ext_vector_type(8)))  __bf16 bf16x8;
typedef __attribute__((ext_vector_type(8)))  float  f32x8;

__device__ __forceinline__ f32x8 wmma_bf16f32(bf16x16 a, bf16x16 b, f32x8 c) {
  // D(16x16 f32) = A(16x32 bf16) * B(32x16 bf16) + C
  return __builtin_amdgcn_wmma_f32_16x16x32_bf16(false, a, false, b, (short)0, c, false, false);
}

// Issue one 16B async global->LDS copy (GVS mode: saddr base + 32-bit vgpr offset).
__device__ __forceinline__ void async_cp16(uint32_t lds_addr, uint32_t voff_bytes,
                                           const void* sbase) {
  asm volatile("global_load_async_to_lds_b128 %0, %1, %2 offset:0"
               :: "v"(lds_addr), "v"(voff_bytes), "s"(sbase) : "memory");
}

// ---------------- f32 -> bf16 ----------------
__global__ __launch_bounds__(256) void cvt_f32_to_bf16(const float* __restrict__ in,
                                                       __bf16* __restrict__ out, int n) {
  int i = blockIdx.x * 256 + threadIdx.x;
  if (i < n) out[i] = (__bf16)in[i];
}

// ---------------- GEMM: C[M,N] = A[M,K] * W[N,K]^T (all row-major) ----------------
// Block = 8 waves = 128x64 tile; the 64x32 W tile (shared by all 8 waves) is staged
// in LDS via double-buffered async copies; A fragments stream direct from global.
__global__ __launch_bounds__(256) void gemm_bf16_xwt(const __bf16* __restrict__ A,
                                                     const __bf16* __restrict__ W,
                                                     float* __restrict__ C,
                                                     int N, int K) {
  // rows padded to 40 bf16 (20 dwords) -> 16-lane column walk hits 16 distinct banks
  __shared__ __align__(16) __bf16 ldsB[2][64][40];

  const int tid = threadIdx.x;
  const int wv = tid >> 5;
  const int lane = tid & 31;
  const int lm = lane & 15, lh = lane >> 4;
  const int m0 = blockIdx.y * 128 + wv * 16;
  const int n0 = blockIdx.x * 64;

  // this thread's 16B chunk of the 64x32 bf16 B tile
  const int crow = tid >> 2;  // 0..63 (tile row = n within tile)
  const int cseg = tid & 3;   // 0..3  (16B segment within the 64B row)
  const uint32_t vbase = (uint32_t)(((n0 + crow) * K + cseg * 8) * 2);  // byte offset in W
  const uint32_t l0 = (uint32_t)(uintptr_t)&ldsB[0][crow][cseg * 8];
  const uint32_t l1 = (uint32_t)(uintptr_t)&ldsB[1][crow][cseg * 8];

  f32x8 acc[4];
#pragma unroll
  for (int nt = 0; nt < 4; ++nt)
#pragma unroll
    for (int v = 0; v < 8; ++v) acc[nt][v] = 0.0f;

  const __bf16* arow = A + (size_t)(m0 + lm) * K;

  auto compute_step = [&](int k0c, int bufc) {
    // A frag: lane(lm) = row; elems 0-15 -> k0+16*lh..k0+16*lh+15
    bf16x8 alo = *(const bf16x8*)(arow + k0c + 16 * lh);
    bf16x8 ahi = *(const bf16x8*)(arow + k0c + 16 * lh + 8);
    bf16x16 a;
#pragma unroll
    for (int i = 0; i < 8; ++i) { a[i] = alo[i]; a[i + 8] = ahi[i]; }
    // preload all 4 B frags (one ds clause, single wait), then 4 back-to-back WMMAs
    const __bf16* bb = &ldsB[bufc][0][0];
    bf16x16 bfr[4];
#pragma unroll
    for (int nt = 0; nt < 4; ++nt)
      bfr[nt] = *(const bf16x16*)(bb + (nt * 16 + lm) * 40 + 16 * lh);
#pragma unroll
    for (int nt = 0; nt < 4; ++nt)
      acc[nt] = wmma_bf16f32(a, bfr[nt], acc[nt]);
  };

  // prologue: stage first tile
  async_cp16(l0, vbase, W);

  int buf = 0;
  int k0 = 0;
  for (; k0 < K - 32; k0 += 32) {
    // steady state: prefetch next tile, wait for current, compute
    async_cp16(buf ? l0 : l1, vbase + (uint32_t)(k0 + 32) * 2, W);
    asm volatile("s_wait_asynccnt 0x1" ::: "memory");  // current tile complete
    __syncthreads();  // every wave waited on its own writes -> whole tile visible
    compute_step(k0, buf);
    __syncthreads();  // all reads of this buffer done before it is overwritten
    buf ^= 1;
  }
  // epilogue: last tile
  asm volatile("s_wait_asynccnt 0x0" ::: "memory");
  __syncthreads();
  compute_step(k0, buf);

#pragma unroll
  for (int nt = 0; nt < 4; ++nt)
#pragma unroll
    for (int v = 0; v < 8; ++v)
      C[(size_t)(m0 + 8 * lh + v) * N + n0 + nt * 16 + lm] = acc[nt][v];
}

// ---------------- RMSNorm + RoPE: x f32 [S, H*D] -> out bf16 [H, S, D] ----------------
// one wave per (s, head); lane owns d in {l, l+32, l+64, l+96} (rotate-half pairs stay in-lane)
__global__ __launch_bounds__(256) void rmsnorm_rope(const float* __restrict__ x,
                                                    const float* __restrict__ wnorm,
                                                    const float* __restrict__ cosb,
                                                    const float* __restrict__ sinb,
                                                    __bf16* __restrict__ out, int H) {
  const int lane = threadIdx.x & 31;
  const int gw = (blockIdx.x * blockDim.x + threadIdx.x) >> 5;
  const int s = gw / H;
  const int h = gw % H;
  const float* row = x + ((size_t)s * H + h) * DHEAD;
  float x0 = row[lane], x1 = row[lane + 32], x2 = row[lane + 64], x3 = row[lane + 96];
  float ss = x0 * x0 + x1 * x1 + x2 * x2 + x3 * x3;
#pragma unroll
  for (int m = 16; m >= 1; m >>= 1) ss += __shfl_xor(ss, m);
  const float r = rsqrtf(ss * (1.0f / (float)DHEAD) + 1e-6f);
  x0 *= r * wnorm[lane];      x1 *= r * wnorm[lane + 32];
  x2 *= r * wnorm[lane + 64]; x3 *= r * wnorm[lane + 96];
  const float* c  = cosb + (size_t)s * DHEAD;
  const float* sn = sinb + (size_t)s * DHEAD;
  const float y0 = x0 * c[lane]      - x2 * sn[lane];
  const float y1 = x1 * c[lane + 32] - x3 * sn[lane + 32];
  const float y2 = x2 * c[lane + 64] + x0 * sn[lane + 64];
  const float y3 = x3 * c[lane + 96] + x1 * sn[lane + 96];
  __bf16* o = out + ((size_t)h * S_LEN + s) * DHEAD;
  o[lane] = (__bf16)y0; o[lane + 32] = (__bf16)y1;
  o[lane + 64] = (__bf16)y2; o[lane + 96] = (__bf16)y3;
}

// ---------------- V: f32 [S, HKV*D] -> bf16 [HKV, D, S] (transposed for PV B-frags) ----
__global__ __launch_bounds__(256) void v_transpose(const float* __restrict__ v,
                                                   __bf16* __restrict__ vt) {
  const size_t o = (size_t)blockIdx.x * 256 + threadIdx.x;  // o = (h*D + d)*S + s
  const size_t total = (size_t)HKV_N * DHEAD * S_LEN;
  if (o >= total) return;
  const int s = (int)(o % S_LEN);
  const size_t hd = o / S_LEN;
  vt[o] = (__bf16)v[(size_t)s * (HKV_N * DHEAD) + hd];
}

// ---------------- Flash-style sliding-window GQA attention ----------------
// one wave per (head hq, 16-query tile); KV steps of 32; out bf16 [S, HQ*D]
__global__ __launch_bounds__(256) void attn_swa(const __bf16* __restrict__ qr,  // [HQ,S,D]
                                                const __bf16* __restrict__ kr,  // [HKV,S,D]
                                                const __bf16* __restrict__ vt,  // [HKV,D,S]
                                                __bf16* __restrict__ outbf) {   // [S,HQ*D]
  __shared__ __align__(16) __bf16 lds[8][16][40];  // per-wave 16x32 P tile, padded stride 40
  const int wv = threadIdx.x >> 5;
  const int lane = threadIdx.x & 31;
  const int lm = lane & 15, lh = lane >> 4;
  const int gw = blockIdx.x * 8 + wv;
  const int nqt = S_LEN / 16;
  const int hq = gw / nqt;
  const int qt = gw % nqt;
  const int hkv = hq >> 1;  // G = HQ/HKV = 2
  const int q0 = qt * 16;
  const float scal = 0.08838834764831845f;  // 1/sqrt(128)

  // Q A-fragments for all 4 K-steps of D=128
  bf16x16 aQ[4];
  const __bf16* qbase = qr + ((size_t)hq * S_LEN + q0 + lm) * DHEAD;
#pragma unroll
  for (int kk = 0; kk < 4; ++kk) {
    bf16x8 lo = *(const bf16x8*)(qbase + kk * 32 + 16 * lh);
    bf16x8 hi = *(const bf16x8*)(qbase + kk * 32 + 16 * lh + 8);
#pragma unroll
    for (int i = 0; i < 8; ++i) { aQ[kk][i] = lo[i]; aQ[kk][i + 8] = hi[i]; }
  }

  f32x8 o[8];
  float mrun[8], lrun[8];
#pragma unroll
  for (int nt = 0; nt < 8; ++nt)
#pragma unroll
    for (int v = 0; v < 8; ++v) o[nt][v] = 0.0f;
#pragma unroll
  for (int v = 0; v < 8; ++v) { mrun[v] = -1e30f; lrun[v] = 0.0f; }

  int jlo = q0 - (WIN - 1); if (jlo < 0) jlo = 0; jlo &= ~31;
  const int jhi = q0 + 15;

  for (int jb = jlo; jb <= jhi; jb += 32) {
    f32x8 s0, s1;
#pragma unroll
    for (int v = 0; v < 8; ++v) { s0[v] = 0.0f; s1[v] = 0.0f; }
    const __bf16* kb = kr + ((size_t)hkv * S_LEN + jb) * DHEAD;
#pragma unroll
    for (int kk = 0; kk < 4; ++kk) {
      bf16x16 b0 = *(const bf16x16*)(kb + (size_t)lm * DHEAD + kk * 32 + 16 * lh);
      bf16x16 b1 = *(const bf16x16*)(kb + (size_t)(16 + lm) * DHEAD + kk * 32 + 16 * lh);
      s0 = wmma_bf16f32(aQ[kk], b0, s0);
      s1 = wmma_bf16f32(aQ[kk], b1, s1);
    }
    // online softmax: row m = 8*lh + v lives in element v across the 16 lanes of half lh
#pragma unroll
    for (int v = 0; v < 8; ++v) {
      const int i = q0 + 8 * lh + v;
      const int j0 = jb + lm, j1 = jb + 16 + lm;
      const bool ok0 = (j0 <= i) && (j0 > i - WIN);
      const bool ok1 = (j1 <= i) && (j1 > i - WIN);
      float a0 = ok0 ? s0[v] * scal : -1e9f;
      float a1 = ok1 ? s1[v] * scal : -1e9f;
      float t = fmaxf(a0, a1);
      t = fmaxf(t, __shfl_xor(t, 1));
      t = fmaxf(t, __shfl_xor(t, 2));
      t = fmaxf(t, __shfl_xor(t, 4));
      t = fmaxf(t, __shfl_xor(t, 8));
      const float mnew = fmaxf(mrun[v], t);
      const float corr = __expf(mrun[v] - mnew);
      const float p0 = ok0 ? __expf(a0 - mnew) : 0.0f;  // explicit zeroing: safe for
      const float p1 = ok1 ? __expf(a1 - mnew) : 0.0f;  // fully-masked steps
      float rs = p0 + p1;
      rs += __shfl_xor(rs, 1);
      rs += __shfl_xor(rs, 2);
      rs += __shfl_xor(rs, 4);
      rs += __shfl_xor(rs, 8);
      lrun[v] = lrun[v] * corr + rs;
      mrun[v] = mnew;
#pragma unroll
      for (int nt = 0; nt < 8; ++nt) o[nt][v] *= corr;
      lds[wv][8 * lh + v][lm] = (__bf16)p0;       // P[m][kv 0..15]
      lds[wv][8 * lh + v][16 + lm] = (__bf16)p1;  // P[m][kv 16..31]
    }
    asm volatile("s_wait_dscnt 0" ::: "memory");  // in-wave LDS transpose round-trip
    bf16x16 aP;
    {
      bf16x8 lo = *(const bf16x8*)&lds[wv][lm][16 * lh];
      bf16x8 hi = *(const bf16x8*)&lds[wv][lm][16 * lh + 8];
#pragma unroll
      for (int i = 0; i < 8; ++i) { aP[i] = lo[i]; aP[i + 8] = hi[i]; }
    }
#pragma unroll
    for (int nt = 0; nt < 8; ++nt) {
      bf16x16 bv = *(const bf16x16*)(vt + ((size_t)hkv * DHEAD + nt * 16 + lm) * S_LEN + jb + 16 * lh);
      o[nt] = wmma_bf16f32(aP, bv, o[nt]);
    }
  }
#pragma unroll
  for (int v = 0; v < 8; ++v) {
    const float inv = 1.0f / lrun[v];
    const int srow = q0 + 8 * lh + v;
#pragma unroll
    for (int nt = 0; nt < 8; ++nt)
      outbf[(size_t)srow * (HQ_N * DHEAD) + hq * DHEAD + nt * 16 + lm] = (__bf16)(o[nt][v] * inv);
  }
}

extern "C" void kernel_launch(void* const* d_in, const int* in_sizes, int n_in,
                              void* d_out, int out_size, void* d_ws, size_t ws_size,
                              hipStream_t stream) {
  (void)in_sizes; (void)n_in; (void)out_size; (void)ws_size;
  const float* hidden = (const float*)d_in[0];
  const float* cosb   = (const float*)d_in[1];
  const float* sinb   = (const float*)d_in[2];
  const float* Wq     = (const float*)d_in[3];
  const float* Wk     = (const float*)d_in[4];
  const float* Wv     = (const float*)d_in[5];
  const float* Wo     = (const float*)d_in[6];
  const float* qnw    = (const float*)d_in[7];
  const float* knw    = (const float*)d_in[8];
  float* out = (float*)d_out;

  char* base = (char*)d_ws;
  size_t off = 0;
  auto take = [&](size_t bytes) -> void* {
    void* p = base + off;
    off += (bytes + 255) & ~(size_t)255;
    return p;
  };
  __bf16* Xbf   = (__bf16*)take((size_t)S_LEN * HID_N * 2);
  __bf16* Wqbf  = (__bf16*)take((size_t)HQ_N * DHEAD * HID_N * 2);
  __bf16* Wkbf  = (__bf16*)take((size_t)HKV_N * DHEAD * HID_N * 2);
  __bf16* Wvbf  = (__bf16*)take((size_t)HKV_N * DHEAD * HID_N * 2);
  __bf16* Wobf  = (__bf16*)take((size_t)HID_N * HQ_N * DHEAD * 2);
  float*  qf    = (float*)take((size_t)S_LEN * HQ_N * DHEAD * 4);
  float*  kf    = (float*)take((size_t)S_LEN * HKV_N * DHEAD * 4);
  float*  vf    = (float*)take((size_t)S_LEN * HKV_N * DHEAD * 4);
  __bf16* qrb   = (__bf16*)take((size_t)HQ_N * S_LEN * DHEAD * 2);
  __bf16* krb   = (__bf16*)take((size_t)HKV_N * S_LEN * DHEAD * 2);
  __bf16* vtb   = (__bf16*)take((size_t)HKV_N * DHEAD * S_LEN * 2);
  __bf16* attnb = (__bf16*)take((size_t)S_LEN * HQ_N * DHEAD * 2);

  auto cdiv = [](int a, int b) { return (a + b - 1) / b; };

  // 1) f32 -> bf16 conversions
  cvt_f32_to_bf16<<<cdiv(S_LEN * HID_N, 256), 256, 0, stream>>>(hidden, Xbf, S_LEN * HID_N);
  cvt_f32_to_bf16<<<cdiv(HQ_N * DHEAD * HID_N, 256), 256, 0, stream>>>(Wq, Wqbf, HQ_N * DHEAD * HID_N);
  cvt_f32_to_bf16<<<cdiv(HKV_N * DHEAD * HID_N, 256), 256, 0, stream>>>(Wk, Wkbf, HKV_N * DHEAD * HID_N);
  cvt_f32_to_bf16<<<cdiv(HKV_N * DHEAD * HID_N, 256), 256, 0, stream>>>(Wv, Wvbf, HKV_N * DHEAD * HID_N);
  cvt_f32_to_bf16<<<cdiv(HID_N * HQ_N * DHEAD, 256), 256, 0, stream>>>(Wo, Wobf, HID_N * HQ_N * DHEAD);

  // 2) QKV projections (WMMA GEMM, X * W^T, async-LDS staged W tiles)
  gemm_bf16_xwt<<<dim3((HQ_N * DHEAD) / 64, S_LEN / 128), 256, 0, stream>>>(Xbf, Wqbf, qf, HQ_N * DHEAD, HID_N);
  gemm_bf16_xwt<<<dim3((HKV_N * DHEAD) / 64, S_LEN / 128), 256, 0, stream>>>(Xbf, Wkbf, kf, HKV_N * DHEAD, HID_N);
  gemm_bf16_xwt<<<dim3((HKV_N * DHEAD) / 64, S_LEN / 128), 256, 0, stream>>>(Xbf, Wvbf, vf, HKV_N * DHEAD, HID_N);

  // 3) RMSNorm + RoPE (q, k) and V transpose
  rmsnorm_rope<<<(S_LEN * HQ_N) / 8, 256, 0, stream>>>(qf, qnw, cosb, sinb, qrb, HQ_N);
  rmsnorm_rope<<<(S_LEN * HKV_N) / 8, 256, 0, stream>>>(kf, knw, cosb, sinb, krb, HKV_N);
  v_transpose<<<cdiv(HKV_N * DHEAD * S_LEN, 256), 256, 0, stream>>>(vf, vtb);

  // 4) sliding-window attention
  attn_swa<<<(HQ_N * (S_LEN / 16)) / 8, 256, 0, stream>>>(qrb, krb, vtb, attnb);

  // 5) output projection -> f32 d_out
  gemm_bf16_xwt<<<dim3(HID_N / 64, S_LEN / 128), 256, 0, stream>>>(attnb, Wobf, out, HID_N, HQ_N * DHEAD);
}
